// GeneratePrediction_68384469287513
// MI455X (gfx1250) — matrix operand order repaired and checked by reading, not compiled, not run on hardware
//
#include <hip/hip_runtime.h>
#include <hip/hip_bf16.h>
#include <math.h>

typedef __attribute__((ext_vector_type(16))) _Float16 v16h;
typedef __attribute__((ext_vector_type(8)))  _Float16 v8h;
typedef __attribute__((ext_vector_type(8)))  float    v8f;
typedef __attribute__((ext_vector_type(4)))  int      i4;

#define B_N   4096
#define C_N   1000
#define C_P   1024
#define V_N   8000
#define V_W   250     // 8000/32 bitmap words
#define V_WP  256     // padded words per row
#define TOPK  5
#define THRESH 0.8f
#define SCALE  0.1f

union F16x16 { v16h v; v8h h[2]; };

#if __has_builtin(__builtin_amdgcn_global_load_async_to_lds_b128)
#define HAVE_ASYNC_LDS 1
typedef __attribute__((address_space(1))) i4* glb_i4p;   // printed as "int4 __device__ *"
typedef __attribute__((address_space(3))) i4* lds_i4p;   // "int4 __shared__ *"
__device__ __forceinline__ void async_copy16(const void* g, void* l) {
  __builtin_amdgcn_global_load_async_to_lds_b128((glb_i4p)(void*)g, (lds_i4p)l, 0, 0);
}
__device__ __forceinline__ void async_wait0() {
#if __has_builtin(__builtin_amdgcn_s_wait_asynccnt)
  __builtin_amdgcn_s_wait_asynccnt(0);
#else
  asm volatile("s_wait_asynccnt 0" ::: "memory");
#endif
}
#else
#define HAVE_ASYNC_LDS 0
#endif

// ---------------- stage 1: sigmoid + threshold mask -> f16 A matrix ----------------
__global__ void k_sigmoid_mask(const float* __restrict__ logits,
                               float* __restrict__ scores,
                               _Float16* __restrict__ masked) {
  int idx = blockIdx.x * 256 + threadIdx.x;      // B_N * C_P threads
  int b = idx >> 10, c = idx & 1023;
  float s = 0.f; _Float16 m = (_Float16)0.f;
  if (c < C_N) {
    float x = logits[(size_t)b * C_N + c];
    s = 1.f / (1.f + __expf(-x));
    if (s > THRESH) m = (_Float16)s;
  }
  scores[idx] = s;
  masked[idx] = m;
}

// ---------------- stage 2: co_occ -> transposed f16 (B^T for WMMA) ----------------
__global__ void k_co_transpose(const float* __restrict__ co, _Float16* __restrict__ coT) {
  int idx = blockIdx.x * 256 + threadIdx.x;      // C_P * C_P
  int j = idx >> 10, k = idx & 1023;
  _Float16 v = (_Float16)0.f;
  if (j < C_N && k < C_N) v = (_Float16)co[(size_t)k * C_N + j];
  coT[idx] = v;
}

// ---------------- stage 3: bit-pack the char matrix ----------------
__global__ void k_bitpack(const int* __restrict__ mat, unsigned* __restrict__ bits) {
  int idx = blockIdx.x * 256 + threadIdx.x;      // C_N * V_WP
  int c = idx >> 8, w = idx & 255;
  unsigned v = 0u;
  if (w < V_W) {
    const int* p = mat + (size_t)c * V_N + w * 32;
    #pragma unroll
    for (int b = 0; b < 32; ++b) v |= (unsigned)(p[b] & 1) << b;
  }
  bits[idx] = v;
}

// ---------------- stage 4: WMMA GEMM  scores += 0.1 * masked @ co_occ ----------------
__global__ void __launch_bounds__(256)
k_adjust_gemm(const _Float16* __restrict__ Amat, const _Float16* __restrict__ BT,
              float* __restrict__ scores) {
  __shared__ __align__(16) _Float16 Al[128 * 40];   // 128x32 tile, stride 40 halves
  __shared__ __align__(16) _Float16 Bl[64 * 40];    // 64x32 B^T tile
  const int t    = threadIdx.x;
  const int lane = t & 31;                          // wave32
  const int wv   = t >> 5;                          // 8 waves -> 8 M-tiles
  const int bm   = blockIdx.x * 128;
  const int bn   = blockIdx.y * 64;
  const int sel  = (lane >> 4) & 1;                 // upper half-wave flag
  v8f acc[4] = {};

  // per-thread tile-copy coordinates (16B granules)
  const int r0 = t >> 2,         c0 = (t & 3) * 8;         // A rows 0..63
  const int r1 = (t + 256) >> 2, c1 = (t & 3) * 8;         // A rows 64..127
  const int rb = t >> 2,         cb = (t & 3) * 8;         // B^T rows 0..63

  for (int kk = 0; kk < C_P; kk += 32) {
    const _Float16* ga0 = Amat + (size_t)(bm + r0) * C_P + kk + c0;
    const _Float16* ga1 = Amat + (size_t)(bm + r1) * C_P + kk + c1;
    const _Float16* gb  = BT   + (size_t)(bn + rb) * C_P + kk + cb;
#if HAVE_ASYNC_LDS
    async_copy16(ga0, Al + r0 * 40 + c0);      // global_load_async_to_lds_b128
    async_copy16(ga1, Al + r1 * 40 + c1);
    async_copy16(gb,  Bl + rb * 40 + cb);
#else
    uint4 d0 = *(const uint4*)ga0;
    uint4 d1 = *(const uint4*)ga1;
    uint4 db = *(const uint4*)gb;
    *(uint4*)(Al + r0 * 40 + c0) = d0;
    *(uint4*)(Al + r1 * 40 + c1) = d1;
    *(uint4*)(Bl + rb * 40 + cb) = db;
#endif
    if (kk + 32 < C_P) {   // gfx1250 global_prefetch of next K tile
      __builtin_prefetch(ga0 + 32, 0, 1);
      __builtin_prefetch(gb + 32, 0, 1);
    }
#if HAVE_ASYNC_LDS
    async_wait0();
#endif
    __syncthreads();

    // A fragment (16x32 f16 layout: lanes 0-15 K={0..7,16..23}, lanes 16-31 K={8..15,24..31})
    F16x16 af;
    {
      int m = wv * 16 + (lane & 15);
      int base = m * 40 + sel * 8;
      af.h[0] = *(const v8h*)(Al + base);
      af.h[1] = *(const v8h*)(Al + base + 16);
    }
    // issue ALL B fragment loads first, then the 4 WMMAs back-to-back
    F16x16 bf[4];
    #pragma unroll
    for (int nt = 0; nt < 4; ++nt) {
      int n = nt * 16 + (lane & 15);
      int base = n * 40 + sel * 8;
      bf[nt].h[0] = *(const v8h*)(Bl + base);
      bf[nt].h[1] = *(const v8h*)(Bl + base + 16);
    }
    #pragma unroll
    for (int nt = 0; nt < 4; ++nt)
      acc[nt] = __builtin_amdgcn_wmma_f32_16x16x32_f16(false, af.v, false, bf[nt].v,
                                                       (short)0, acc[nt], false, false);
    __syncthreads();
  }

  // D layout: VGPR r, lanes0-15 -> M=r, lanes16-31 -> M=r+8, N=lane&15
  const int rowOff = sel * 8;
  #pragma unroll
  for (int nt = 0; nt < 4; ++nt) {
    int col = bn + nt * 16 + (lane & 15);
    #pragma unroll
    for (int r = 0; r < 8; ++r) {
      int row = bm + wv * 16 + r + rowOff;
      size_t idx = (size_t)row * C_P + col;
      scores[idx] = scores[idx] + SCALE * acc[nt][r];
    }
  }
}

// ---------------- stage 5: softmax + top-5 per row ----------------
__global__ void __launch_bounds__(256)
k_softmax_top5(const float* __restrict__ logits, float* __restrict__ out) {
  __shared__ float vals[V_N];
  __shared__ float redV[256];
  __shared__ int   redI[256];
  const int t = threadIdx.x;
  const int b = blockIdx.x;
  const float* row = logits + (size_t)b * V_N;
  float mx = -INFINITY;
  for (int j = t; j < V_N; j += 256) { float v = row[j]; vals[j] = v; mx = fmaxf(mx, v); }
  redV[t] = mx; __syncthreads();
  for (int s = 128; s > 0; s >>= 1) { if (t < s) redV[t] = fmaxf(redV[t], redV[t + s]); __syncthreads(); }
  const float M = redV[0]; __syncthreads();
  float sum = 0.f;
  for (int j = t; j < V_N; j += 256) sum += __expf(vals[j] - M);
  redV[t] = sum; __syncthreads();
  for (int s = 128; s > 0; s >>= 1) { if (t < s) redV[t] += redV[t + s]; __syncthreads(); }
  const float S = redV[0]; __syncthreads();
  for (int k = 0; k < TOPK; ++k) {
    float bv = -INFINITY; int bi = 0x7fffffff;
    for (int j = t; j < V_N; j += 256) {
      float v = vals[j];
      if (v > bv || (v == bv && j < bi)) { bv = v; bi = j; }
    }
    redV[t] = bv; redI[t] = bi; __syncthreads();
    for (int s = 128; s > 0; s >>= 1) {
      if (t < s && (redV[t + s] > redV[t] ||
                    (redV[t + s] == redV[t] && redI[t + s] < redI[t]))) {
        redV[t] = redV[t + s]; redI[t] = redI[t + s];
      }
      __syncthreads();
    }
    if (t == 0) {
      int idx = redI[0];
      out[(size_t)b * TOPK + k]          = (float)idx;               // class_indices
      out[20480 + (size_t)b * TOPK + k]  = __expf(redV[0] - M) / S;  // class_scores
      vals[idx] = -INFINITY;
    }
    __syncthreads();
  }
}

// ---------------- stage 6: per-sample decode ----------------
__global__ void __launch_bounds__(256)
k_decode(const float* __restrict__ scores, const unsigned* __restrict__ bits,
         float* __restrict__ out) {
  __shared__ float sc[C_P];
  __shared__ float redV[256];
  __shared__ int   redI[256];
  __shared__ int   redC[256];
  __shared__ unsigned hitw[256];
  const int t = threadIdx.x;
  const int b = blockIdx.x;

  for (int j = t; j < C_P; j += 256)
    sc[j] = (j < C_N) ? scores[(size_t)b * C_P + j] : -INFINITY;
  __syncthreads();

  // argmax + selected-count
  float bv = -INFINITY; int bi = 0x7fffffff; int nsel = 0;
  for (int j = t; j < C_P; j += 256) {
    float v = sc[j];
    if (v > bv || (v == bv && j < bi)) { bv = v; bi = j; }
    if (v > THRESH) nsel++;
  }
  redV[t] = bv; redI[t] = bi; redC[t] = nsel; __syncthreads();
  for (int s = 128; s > 0; s >>= 1) {
    if (t < s) {
      if (redV[t + s] > redV[t] || (redV[t + s] == redV[t] && redI[t + s] < redI[t])) {
        redV[t] = redV[t + s]; redI[t] = redI[t + s];
      }
      redC[t] += redC[t + s];
    }
    __syncthreads();
  }
  const int   max_idx   = redI[0];
  const float max_score = redV[0];
  const int   num_sel   = redC[0];
  __syncthreads();
  if (t == 0) sc[max_idx] = -INFINITY;

  unsigned w = (t < V_W) ? bits[(size_t)max_idx * V_WP + t] : 0u;
  unsigned prevW = w;
  redC[t] = __popc(w); __syncthreads();
  for (int s = 128; s > 0; s >>= 1) { if (t < s) redC[t] += redC[t + s]; __syncthreads(); }
  int cnt = redC[0];
  __syncthreads();

  float s0 = (num_sel > 0) ? max_score : -INFINITY;  // sorted_scores[0] (masked)
  float s_im1 = s0, s_im2 = s0;
  int i = 1;
  while (i < num_sel && cnt > 1) {
    float cv = -INFINITY; int ci = 0x7fffffff;
    for (int j = t; j < C_P; j += 256) {
      float v = sc[j];
      if (v > cv || (v == cv && j < ci)) { cv = v; ci = j; }
    }
    redV[t] = cv; redI[t] = ci; __syncthreads();
    for (int s = 128; s > 0; s >>= 1) {
      if (t < s && (redV[t + s] > redV[t] ||
                    (redV[t + s] == redV[t] && redI[t + s] < redI[t]))) {
        redV[t] = redV[t + s]; redI[t] = redI[t + s];
      }
      __syncthreads();
    }
    int idx2 = redI[0]; float val2 = redV[0];
    __syncthreads();
    if (t == 0) sc[idx2] = -INFINITY;
    s_im2 = s_im1; s_im1 = val2;
    prevW = w;
    unsigned rw = (t < V_W) ? bits[(size_t)idx2 * V_WP + t] : 0u;
    w &= rw;
    redC[t] = __popc(w); __syncthreads();
    for (int s = 128; s > 0; s >>= 1) { if (t < s) redC[t] += redC[t + s]; __syncthreads(); }
    cnt = redC[0];
    ++i;
    __syncthreads();
  }
  bool empty = (cnt == 0);
  if (empty) { --i; w = prevW; }
  float selS = empty ? s_im2 : s_im1;
  float hit_score = (i == 1) ? max_score : selS;

  hitw[t] = w; __syncthreads();
  if (t == 0) {
    int ids[TOPK]; int nh = 0;
    for (int wd = 0; wd < V_W && nh < TOPK; ++wd) {
      unsigned x = hitw[wd];
      while (x && nh < TOPK) {
        int bpos = __ffs(x) - 1;
        ids[nh++] = wd * 32 + bpos;
        x &= x - 1;
      }
    }
    int num_hit = nh;
    for (int k = nh; k < TOPK; ++k) ids[k] = -1;
    for (int k = 0; k < TOPK; ++k) out[40960 + (size_t)b * TOPK + k] = (float)ids[k];
    out[61440 + b] = hit_score;
    float cls0 = out[(size_t)b * TOPK];
    float cs0  = out[20480 + (size_t)b * TOPK];
    out[65536 + b] = (num_hit == 1) ? (float)ids[0] : cls0;
    out[69632 + b] = (cs0 < 0.85f && num_hit == 1) ? (float)ids[0] : cls0;
  }
}

extern "C" void kernel_launch(void* const* d_in, const int* in_sizes, int n_in,
                              void* d_out, int out_size, void* d_ws, size_t ws_size,
                              hipStream_t stream) {
  const float* class_logits = (const float*)d_in[0];
  const float* compo_logits = (const float*)d_in[1];
  const float* co_occ       = (const float*)d_in[2];
  const int*   matrix       = (const int*)d_in[3];
  float* out = (float*)d_out;
  char*  ws  = (char*)d_ws;
  _Float16* maskedF16 = (_Float16*)(ws);                        //  8 MB  [4096,1024] f16
  _Float16* coT       = (_Float16*)(ws + (8u  << 20));          //  2 MB  [1024,1024] f16
  unsigned* bits      = (unsigned*)(ws + (10u << 20));          //  1 MB  [1000,256] u32
  float*    scores    = (float*)   (ws + (11u << 20));          // 16 MB  [4096,1024] f32

  k_sigmoid_mask<<<(B_N * C_P) / 256, 256, 0, stream>>>(compo_logits, scores, maskedF16);
  k_co_transpose<<<(C_P * C_P) / 256, 256, 0, stream>>>(co_occ, coT);
  k_bitpack<<<(C_N * V_WP) / 256, 256, 0, stream>>>(matrix, bits);
  dim3 g(B_N / 128, C_P / 64);
  k_adjust_gemm<<<g, 256, 0, stream>>>(maskedF16, coT, scores);
  k_softmax_top5<<<B_N, 256, 0, stream>>>(class_logits, out);
  k_decode<<<B_N, 256, 0, stream>>>(scores, bits, out);
}
